// BoundaryPredictor3_78391743087042
// MI455X (gfx1250) — matrix operand order repaired and checked by reading, not compiled, not run on hardware
//
#include <hip/hip_runtime.h>
#include <math.h>

// ---- problem constants (B=4, L=1024, D=512, H=8, HD=64) ----
#define BB 4
#define LL 1024
#define DD 512
#define HH 8
#define HDD 64
#define NROWS (BB*LL)          // 4096
#define EPSF 1e-8f
#define LN_EPSF 1e-5f

// B-tile staging: 64 N-rows x 128 K elems per chunk, padded to 136 elems/row
#define BK_CHUNK 128
#define BROW_PAD 136
#define NCHUNK   (DD / BK_CHUNK)   // 4

typedef __bf16 bf16_t;
typedef __attribute__((ext_vector_type(16))) __bf16 v16bf;
typedef __attribute__((ext_vector_type(8)))  __bf16 v8bf;
typedef __attribute__((ext_vector_type(8)))  float   v8f;

typedef unsigned int u32x4 __attribute__((ext_vector_type(4)));
typedef int          i32x4 __attribute__((ext_vector_type(4)));
typedef int          i32x8 __attribute__((ext_vector_type(8)));

#if __has_builtin(__builtin_amdgcn_tensor_load_to_lds) && __has_builtin(__builtin_amdgcn_s_wait_tensorcnt)
#define BP3_HAVE_TDM 1
#endif

// ------------------------------------------------------------------
// f32 -> bf16 convert (weights)
// ------------------------------------------------------------------
__global__ void bp3_cvt_bf16(const float* __restrict__ src, bf16_t* __restrict__ dst, int n) {
    int i = blockIdx.x * blockDim.x + threadIdx.x;
    if (i < n) dst[i] = (bf16_t)src[i];
}

// ------------------------------------------------------------------
// Prep: per row (b,l): l2-normalize (store f32 + bf16), LayerNorm (f32),
// per-head logits  lq_h . hn_h / 8.   256 threads, each owns d=t and d=t+256.
// ------------------------------------------------------------------
__global__ void bp3_prep(const float* __restrict__ hidden,
                         const float* __restrict__ gamma,
                         const float* __restrict__ beta,
                         const float* __restrict__ lq,
                         bf16_t* __restrict__ normbf,
                         float*  __restrict__ norml,
                         float*  __restrict__ hn,
                         float*  __restrict__ logits) {
    const int row = blockIdx.x;                  // b*L + l
    const int t   = threadIdx.x;
    const float* x = hidden + (size_t)row * DD;
    const float x0 = x[t], x1 = x[t + 256];

    __shared__ float red[256];
    // sum
    red[t] = x0 + x1; __syncthreads();
    for (int o = 128; o > 0; o >>= 1) { if (t < o) red[t] += red[t + o]; __syncthreads(); }
    const float mu = red[0] * (1.0f / (float)DD);
    __syncthreads();
    // sum of squares
    red[t] = x0 * x0 + x1 * x1; __syncthreads();
    for (int o = 128; o > 0; o >>= 1) { if (t < o) red[t] += red[t + o]; __syncthreads(); }
    const float sumsq = red[0];
    __syncthreads();

    const float var  = sumsq * (1.0f / (float)DD) - mu * mu;
    const float rstd = rsqrtf(var + LN_EPSF);
    const float inv  = 1.0f / fmaxf(sqrtf(sumsq), EPSF);

    const float n0 = x0 * inv, n1 = x1 * inv;
    norml[(size_t)row * DD + t]        = n0;
    norml[(size_t)row * DD + t + 256]  = n1;
    normbf[(size_t)row * DD + t]       = (bf16_t)n0;
    normbf[(size_t)row * DD + t + 256] = (bf16_t)n1;

    const float h0 = (x0 - mu) * rstd * gamma[t]       + beta[t];
    const float h1 = (x1 - mu) * rstd * gamma[t + 256] + beta[t + 256];
    hn[(size_t)row * DD + t]       = h0;
    hn[(size_t)row * DD + t + 256] = h1;

    // logits: segmented reduce in groups of 64 threads (one head each), heads 0-3 then 4-7
    red[t] = lq[t] * h0; __syncthreads();
    for (int o = 32; o > 0; o >>= 1) { if ((t & 63) < o) red[t] += red[t + o]; __syncthreads(); }
    if ((t & 63) == 0) logits[(size_t)row * HH + (t >> 6)] = red[t] * 0.125f;   // HD^-0.5 = 1/8
    __syncthreads();
    red[t] = lq[t + 256] * h1; __syncthreads();
    for (int o = 32; o > 0; o >>= 1) { if ((t & 63) < o) red[t] += red[t + o]; __syncthreads(); }
    if ((t & 63) == 0) logits[(size_t)row * HH + 4 + (t >> 6)] = red[t] * 0.125f;
}

// ------------------------------------------------------------------
// TDM: DMA one 64 x BK_CHUNK bf16 tile of W (row stride DD elems) into LDS,
// inserting 16B of padding after every 256B row (-> BROW_PAD layout).
// Issued by one wave; tracked with TENSORcnt (in-order completion).
// ------------------------------------------------------------------
#ifdef BP3_HAVE_TDM
__device__ __forceinline__ void bp3_tdm_load_b(const bf16_t* gsrc, unsigned int lds_off) {
    const unsigned long long ga = (unsigned long long)(uintptr_t)gsrc;
    u32x4 g0;
    g0[0] = 1u;                                            // count=1 (valid user descriptor)
    g0[1] = lds_off;                                       // lds_addr (bytes)
    g0[2] = (unsigned int)ga;                              // global_addr[31:0]
    g0[3] = (unsigned int)((ga >> 32) & 0x01FFFFFFu)       // global_addr[56:32]
          | (2u << 30);                                    // type = 2 ("image")
    i32x8 g1;
    g1[0] = (int)(0x00010000u        // data_size=1 -> 2 bytes
          | (1u << 20)               // pad_enable
          | (5u << 22)               // pad_interval: 64 DWORDs (=256B, one row)
          | (3u << 25));             // pad_amount: 4 DWORDs (=16B)
    g1[1] = (int)(((unsigned)DD & 0xFFFFu) << 16);         // tensor_dim0[15:0] = 512
    g1[2] = (int)((((unsigned)DD >> 16) & 0xFFFFu)         // tensor_dim0[31:16]
          | ((unsigned)DD << 16));                         // tensor_dim1[15:0] = 512
    g1[3] = (int)(((unsigned)BK_CHUNK & 0xFFFFu) << 16);   // tile_dim0 = 128
    g1[4] = 64;                                            // tile_dim1 = 64 rows, tile_dim2 = 0
    g1[5] = DD;                                            // tensor_dim0_stride[31:0] = 512
    g1[6] = 0;                                             // stride hi / dim1_stride (unused, 2D)
    g1[7] = 0;
    i32x4 gz = {0, 0, 0, 0};
#if defined(__clang_major__) && (__clang_major__ >= 23)
    i32x8 g5 = {0, 0, 0, 0, 0, 0, 0, 0};
    __builtin_amdgcn_tensor_load_to_lds(g0, g1, gz, gz, g5, 0);
#else
    __builtin_amdgcn_tensor_load_to_lds(g0, g1, gz, gz, 0);
#endif
}
#endif

// ------------------------------------------------------------------
// bf16 WMMA GEMM:  out[n,j] = sum_k X[n,k] * W[j,k] + bias[j]   (X: 4096xD, W: DxD)
// MODE 0: gelu(.) -> bf16 out     MODE 1: (.) + res -> f32 out
// Block = 8 waves; block tile 128(M) x 64(N); wave tile 16 x 64.
// B strip staged in LDS via double-buffered TDM (4 chunks of 64x128);
// DMA of chunk h+1 overlaps compute of chunk h. A streamed from global.
// ------------------------------------------------------------------
template<int MODE>
__global__ __launch_bounds__(256)
void bp3_gemm(const bf16_t* __restrict__ Xb, const bf16_t* __restrict__ Wb,
              const float* __restrict__ bias, const float* __restrict__ res,
              bf16_t* __restrict__ outb, float* __restrict__ outf) {
    __shared__ bf16_t ldsB[2][64][BROW_PAD];  // 2 x 17 KB

    const int lane = threadIdx.x & 31;
    const int wv   = threadIdx.x >> 5;
    const int m0   = blockIdx.x * 128 + wv * 16;
    const int n0   = blockIdx.y * 64;
    const int lg   = lane >> 4;               // lane group 0/1
    const int ll   = lane & 15;

    v8f acc[4];
#pragma unroll
    for (int nt = 0; nt < 4; ++nt)
#pragma unroll
        for (int i = 0; i < 8; ++i) acc[nt][i] = 0.0f;

    const size_t arow = (size_t)(m0 + ll) * DD;
    const bf16_t* wbase = Wb + (size_t)n0 * DD;

#ifdef BP3_HAVE_TDM
    if (wv == 0) {
        bp3_tdm_load_b(wbase,            (unsigned int)(uintptr_t)&ldsB[0][0][0]);
        bp3_tdm_load_b(wbase + BK_CHUNK, (unsigned int)(uintptr_t)&ldsB[1][0][0]);
    }
#endif

    for (int h = 0; h < NCHUNK; ++h) {
        const int buf = h & 1;
        const int k0  = h * BK_CHUNK;
#ifdef BP3_HAVE_TDM
        // chunks complete in order: TENSORcnt <= 1 (or 0 at tail) => chunk h resident
        // (builtin needs a literal immediate, hence the branch)
        if (wv == 0) {
            if (h < NCHUNK - 2) __builtin_amdgcn_s_wait_tensorcnt(1);
            else                __builtin_amdgcn_s_wait_tensorcnt(0);
        }
        __syncthreads();
#else
        // cooperative fallback: 64*128 bf16 = 1024 v8bf, 256 threads -> 4 iters
        for (int it = 0; it < 4; ++it) {
            const int idx = it * 256 + threadIdx.x;   // v8bf index
            const int row = idx >> 4;                 // 16 v8bf per row
            const int col = (idx & 15) << 3;
            *(v8bf*)&ldsB[buf][row][col] =
                *(const v8bf*)(Wb + (size_t)(n0 + row) * DD + k0 + col);
        }
        __syncthreads();
#endif
        // ---- K loop over this chunk ----
#pragma unroll
        for (int kk = 0; kk < BK_CHUNK; kk += 32) {
            // A fragment (16x32): lane = row m, K chunks [kb,kb+8) & [kb+16,kb+24)
            const int kbA = k0 + kk + (lg << 3);
            v8bf a0 = *(const v8bf*)(Xb + arow + kbA);
            v8bf a1 = *(const v8bf*)(Xb + arow + kbA + 16);

            // Pre-load ALL four B fragments so ds_loads clause together and the
            // WMMA chain isn't serialized on per-fragment dscnt==0 waits.
            const int kbB = kk + (lg << 3);
            v16bf bv[4];
#pragma unroll
            for (int nt = 0; nt < 4; ++nt) {
                const bf16_t* bp = &ldsB[buf][nt * 16 + ll][kbB];
                v8bf b0 = *(const v8bf*)bp;
                v8bf b1 = *(const v8bf*)(bp + 16);
                bv[nt] = __builtin_shufflevector(b0, b1, 0,1,2,3,4,5,6,7,8,9,10,11,12,13,14,15);
            }
            v16bf av = __builtin_shufflevector(a0, a1, 0,1,2,3,4,5,6,7,8,9,10,11,12,13,14,15);
#pragma unroll
            for (int nt = 0; nt < 4; ++nt) {
                acc[nt] = __builtin_amdgcn_wmma_f32_16x16x32_bf16(
                              false, av, false, bv[nt], (short)0, acc[nt], false, false);
            }
        }
        __syncthreads();   // all waves done with this buffer
#ifdef BP3_HAVE_TDM
        if (wv == 0 && h + 2 < NCHUNK)
            bp3_tdm_load_b(wbase + (size_t)(h + 2) * BK_CHUNK,
                           (unsigned int)(uintptr_t)&ldsB[buf][0][0]);
#endif
    }

    // epilogue: D-matrix layout, m = m0 + r + 8*lanegroup, n = n0 + nt*16 + ll
#pragma unroll
    for (int nt = 0; nt < 4; ++nt) {
        const int n = n0 + nt * 16 + ll;
        const float bj = bias[n];
#pragma unroll
        for (int r = 0; r < 8; ++r) {
            const int m = m0 + r + (lg << 3);
            float v = acc[nt][r] + bj;
            if (MODE == 0) {
                // exact gelu: 0.5*x*(1+erf(x/sqrt(2)))
                v = 0.5f * v * (1.0f + erff(v * 0.70710678118654752f));
                outb[(size_t)m * DD + n] = (bf16_t)v;
            } else {
                outf[(size_t)m * DD + n] = v + res[(size_t)m * DD + n];
            }
        }
    }
}

// ------------------------------------------------------------------
// Row L2 norms of z
// ------------------------------------------------------------------
__global__ void bp3_rnorm(const float* __restrict__ z, float* __restrict__ rnorm) {
    const int row = blockIdx.x, t = threadIdx.x;
    const float* zr = z + (size_t)row * DD;
    __shared__ float red[256];
    red[t] = zr[t] * zr[t] + zr[t + 256] * zr[t + 256]; __syncthreads();
    for (int o = 128; o > 0; o >>= 1) { if (t < o) red[t] += red[t + o]; __syncthreads(); }
    if (t == 0) rnorm[row] = fmaxf(sqrtf(red[0]), EPSF);
}

// ------------------------------------------------------------------
// probs[b,l] = clip((1 - (cos(z_l, z_{l+1}) + bias))*0.5, 0, 1);  probs[b,L-1] = 0
// ------------------------------------------------------------------
__global__ void bp3_probs(const float* __restrict__ z, const float* __restrict__ rnorm,
                          const float* __restrict__ sim_bias, float* __restrict__ probs) {
    const int row = blockIdx.x, t = threadIdx.x;
    const int l = row & (LL - 1);
    if (l == LL - 1) { if (t == 0) probs[row] = 0.0f; return; }
    const float* z0 = z + (size_t)row * DD;
    const float* z1 = z0 + DD;
    __shared__ float red[256];
    red[t] = z0[t] * z1[t] + z0[t + 256] * z1[t + 256]; __syncthreads();
    for (int o = 128; o > 0; o >>= 1) { if (t < o) red[t] += red[t + o]; __syncthreads(); }
    if (t == 0) {
        const float cs = red[0] / (rnorm[row] * rnorm[row + 1]);
        float p = (1.0f - (cs + sim_bias[0])) * 0.5f;
        probs[row] = fminf(fmaxf(p, 0.0f), 1.0f);
    }
}

// ------------------------------------------------------------------
// Per-batch segmentation: exclusive prefix sum of hard -> seg id; emit
// seg_start[b,s] (first l of segment s) and seg_count[b,s] (#valid l, l < len).
// ------------------------------------------------------------------
__global__ void bp3_segs(const float* __restrict__ probs, const float* __restrict__ lengths,
                         int* __restrict__ seg_start, int* __restrict__ seg_count) {
    const int b = blockIdx.x, l = threadIdx.x;
    __shared__ int hh[LL];
    __shared__ int scan[LL];
    __shared__ int sc[LL];
    __shared__ int ss[LL];
    const int hard = (probs[b * LL + l] > 0.5f) ? 1 : 0;
    hh[l] = hard; scan[l] = hard; sc[l] = 0; ss[l] = 0;
    __syncthreads();
    for (int off = 1; off < LL; off <<= 1) {
        int add = (l >= off) ? scan[l - off] : 0;
        __syncthreads();
        scan[l] += add;
        __syncthreads();
    }
    const int seg = scan[l] - hh[l];                       // exclusive scan
    const int len = (int)(lengths[b] * (float)LL);
    if (l == 0 || hh[l - 1] == 1) ss[seg] = l;             // unique writer per segment
    if (l < len) atomicAdd(&sc[seg], 1);
    __syncthreads();
    seg_start[b * LL + l] = ss[l];
    seg_count[b * LL + l] = sc[l];
}

// ------------------------------------------------------------------
// Segment softmax pooling: one block per (b,s). 8 waves: one head per wave for
// max/sumexp; then 256 threads accumulate the 512 output dims.
// ------------------------------------------------------------------
__global__ void bp3_pool(const float* __restrict__ hn, const float* __restrict__ logits,
                         const int* __restrict__ seg_start, const int* __restrict__ seg_count,
                         float* __restrict__ out) {
    const int bs = blockIdx.x;             // b*L + s
    const int b  = bs >> 10;
    const int t  = threadIdx.x;
    float* o = out + (size_t)bs * DD;
    const int cnt = seg_count[bs];
    if (cnt == 0) { o[t] = 0.0f; o[t + 256] = 0.0f; return; }
    const int start = seg_start[bs];
    const float* lg = logits + ((size_t)b * LL + start) * HH;

    __shared__ float mx[HH], sm[HH];
    const int h = t >> 5, j = t & 31;
    float m = -1e30f;
    for (int i = j; i < cnt; i += 32) m = fmaxf(m, lg[i * HH + h]);
#pragma unroll
    for (int o2 = 16; o2 > 0; o2 >>= 1) m = fmaxf(m, __shfl_xor(m, o2, 32));
    if (j == 0) mx[h] = m;
    __syncthreads();
    const float mm = mx[h];
    float se = 0.0f;
    for (int i = j; i < cnt; i += 32) se += __expf(lg[i * HH + h] - mm);
#pragma unroll
    for (int o2 = 16; o2 > 0; o2 >>= 1) se += __shfl_xor(se, o2, 32);
    if (j == 0) sm[h] = se;
    __syncthreads();

    const int d0 = t, d1 = t + 256;
    const int h0 = d0 >> 6, h1 = d1 >> 6;
    const float m0 = mx[h0], m1 = mx[h1];
    const float i0 = 1.0f / sm[h0], i1 = 1.0f / sm[h1];
    const float* hrow = hn + ((size_t)b * LL + start) * DD;
    float a0 = 0.0f, a1 = 0.0f;
    for (int i = 0; i < cnt; ++i) {
        const float w0 = __expf(lg[i * HH + h0] - m0) * i0;
        const float w1 = __expf(lg[i * HH + h1] - m1) * i1;
        a0 += w0 * hrow[(size_t)i * DD + d0];
        a1 += w1 * hrow[(size_t)i * DD + d1];
    }
    o[d0] = a0;
    o[d1] = a1;
}

// ------------------------------------------------------------------
extern "C" void kernel_launch(void* const* d_in, const int* in_sizes, int n_in,
                              void* d_out, int out_size, void* d_ws, size_t ws_size,
                              hipStream_t stream) {
    const float* hidden   = (const float*)d_in[0];
    const float* lengths  = (const float*)d_in[1];
    const float* W1       = (const float*)d_in[2];
    const float* b1       = (const float*)d_in[3];
    const float* W2       = (const float*)d_in[4];
    const float* b2       = (const float*)d_in[5];
    const float* sim_bias = (const float*)d_in[8];
    const float* lq       = (const float*)d_in[9];
    const float* ln_gamma = (const float*)d_in[13];
    const float* ln_beta  = (const float*)d_in[14];
    float* out = (float*)d_out;

    // workspace carve-up (all offsets 256B-aligned)
    char* w = (char*)d_ws;
    bf16_t* normbf = (bf16_t*)w; w += (size_t)NROWS * DD * 2;   // 4 MB
    float*  norml  = (float*)w;  w += (size_t)NROWS * DD * 4;   // 8 MB
    float*  hn     = (float*)w;  w += (size_t)NROWS * DD * 4;   // 8 MB
    bf16_t* c1bf   = (bf16_t*)w; w += (size_t)NROWS * DD * 2;   // 4 MB
    float*  z      = (float*)w;  w += (size_t)NROWS * DD * 4;   // 8 MB
    bf16_t* w1bf   = (bf16_t*)w; w += (size_t)DD * DD * 2;      // 0.5 MB
    bf16_t* w2bf   = (bf16_t*)w; w += (size_t)DD * DD * 2;      // 0.5 MB
    float*  rnorm  = (float*)w;  w += (size_t)NROWS * 4;
    float*  probs  = (float*)w;  w += (size_t)NROWS * 4;
    int*    sstart = (int*)w;    w += (size_t)NROWS * 4;
    int*    scount = (int*)w;    w += (size_t)NROWS * 4;
    float*  logits = (float*)w;  w += (size_t)NROWS * HH * 4;

    // 1) weights -> bf16
    bp3_cvt_bf16<<<(DD * DD + 255) / 256, 256, 0, stream>>>(W1, w1bf, DD * DD);
    bp3_cvt_bf16<<<(DD * DD + 255) / 256, 256, 0, stream>>>(W2, w2bf, DD * DD);

    // 2) fused l2norm + LayerNorm + head logits
    bp3_prep<<<NROWS, 256, 0, stream>>>(hidden, ln_gamma, ln_beta, lq,
                                        normbf, norml, hn, logits);

    // 3) MLP via WMMA + pipelined TDM weights: gelu(X W1^T + b1) then (. W2^T + b2) + residual
    dim3 ggrid(NROWS / 128, DD / 64);
    bp3_gemm<0><<<ggrid, 256, 0, stream>>>(normbf, w1bf, b1, nullptr, c1bf, nullptr);
    bp3_gemm<1><<<ggrid, 256, 0, stream>>>(c1bf, w2bf, b2, norml, nullptr, z);

    // 4) cosine-sim boundary probabilities
    bp3_rnorm<<<NROWS, 256, 0, stream>>>(z, rnorm);
    bp3_probs<<<NROWS, 256, 0, stream>>>(z, rnorm, sim_bias, probs);

    // 5) segmentation (prefix sum per batch)
    bp3_segs<<<BB, LL, 0, stream>>>(probs, lengths, sstart, scount);

    // 6) per-segment softmax pooling -> output
    bp3_pool<<<NROWS, 256, 0, stream>>>(hn, logits, sstart, scount, out);
    (void)in_sizes; (void)n_in; (void)out_size; (void)ws_size;
}
